// NonLocalLayer_8864812499327
// MI455X (gfx1250) — matrix operand order repaired and checked by reading, not compiled, not run on hardware
//
#include <hip/hip_runtime.h>
#include <hip/hip_bf16.h>

typedef _Float16 f16;
typedef __attribute__((ext_vector_type(16))) _Float16 v16h;
typedef __attribute__((ext_vector_type(8)))  _Float16 v8h;
typedef __attribute__((ext_vector_type(8)))  float    v8f;
typedef __attribute__((ext_vector_type(4)))  float    v4f;

static constexpr int N_   = 4;
static constexpr int CIN  = 256;
static constexpr int CI   = 128;
static constexpr int HW   = 4096;

#define DEVINL static __device__ __forceinline__

DEVINL v8f vzero() {
    v8f z;
#pragma unroll
    for (int i = 0; i < 8; ++i) z[i] = 0.0f;
    return z;
}

DEVINL v16h cat16(v8h a, v8h b) {
    v16h r;
#pragma unroll
    for (int i = 0; i < 8; ++i) { r[i] = a[i]; r[i + 8] = b[i]; }
    return r;
}

// A fragment (16x32 f16, MxK), source row-major [m][k] with leading dim ld (halves).
// ISA layout: lane m=lane&15, half=lane>>4; K runs [8*half,+8) and [16+8*half,+8).
DEVINL v16h load_a16x32(const f16* base, int ld, int m, int half) {
    const v8h* p0 = (const v8h*)(base + (size_t)m * ld + 8 * half);
    const v8h* p1 = (const v8h*)(base + (size_t)m * ld + 16 + 8 * half);
    return cat16(*p0, *p1);
}

// Same, but from fp32 source with on-the-fly f16 conversion.
DEVINL v16h load_a16x32_f32(const float* base, int ld, int m, int half) {
    v4f a0 = *(const v4f*)(base + (size_t)m * ld + 8 * half);
    v4f a1 = *(const v4f*)(base + (size_t)m * ld + 8 * half + 4);
    v4f b0 = *(const v4f*)(base + (size_t)m * ld + 16 + 8 * half);
    v4f b1 = *(const v4f*)(base + (size_t)m * ld + 16 + 8 * half + 4);
    v16h r;
#pragma unroll
    for (int i = 0; i < 4; ++i) {
        r[i] = (f16)a0[i]; r[4 + i] = (f16)a1[i];
        r[8 + i] = (f16)b0[i]; r[12 + i] = (f16)b1[i];
    }
    return r;
}

// B fragment (32x16 f16, KxN): B[k][n] = buf[n*ld + k] ("transposed" buffer, K contiguous).
// ISA layout: N = lane&15; K runs [8*half,+8) and [16+8*half,+8).
DEVINL v16h load_bT(const f16* base, int ld, int n, int half) {
    const v8h* p0 = (const v8h*)(base + (size_t)n * ld + 8 * half);
    const v8h* p1 = (const v8h*)(base + (size_t)n * ld + 16 + 8 * half);
    return cat16(*p0, *p1);
}

DEVINL v8f wmma_f16(v16h a, v16h b, v8f c) {
    return __builtin_amdgcn_wmma_f32_16x16x32_f16(false, a, false, b, (short)0, c,
                                                  false, false);
}

// ---------------------------------------------------------------------------
// Kernel 1: projections.  x[n][256][HW] (fp32) -> thetaT/phiT [n][HW][128] f16
// (pos-major for cheap A/B fragment loads later) and g [n][128][HW] f16.
// Block: 256 thr, p-block = 64 positions, stage x tile transposed in LDS.
// ---------------------------------------------------------------------------
__global__ void __launch_bounds__(256) proj_kernel(
    const float* __restrict__ x,
    const float* __restrict__ th_w, const float* __restrict__ th_b,
    const float* __restrict__ ph_w, const float* __restrict__ ph_b,
    const float* __restrict__ gg_w, const float* __restrict__ gg_b,
    f16* __restrict__ thetaT, f16* __restrict__ phiT, f16* __restrict__ gH)
{
    constexpr int LDX = 264;              // 256 + pad, keeps 16B alignment (528B rows)
    __shared__ f16 xT[64 * LDX];          // [p][c] f16, 33.8 KB

    const int bx    = blockIdx.x;
    const int batch = bx >> 6;            // 64 p-blocks per batch
    const int p0    = (bx & 63) * 64;
    const float* xb = x + (size_t)batch * CIN * HW;

    const int tid = threadIdx.x;
    // Stage: group g=tid>>4 walks c rows, lane16 loads 4 consecutive p.
    const int lane16 = tid & 15, grp = tid >> 4;
    for (int c = grp; c < CIN; c += 16) {
        v4f v = *(const v4f*)(xb + (size_t)c * HW + p0 + lane16 * 4);
#pragma unroll
        for (int q = 0; q < 4; ++q) xT[(lane16 * 4 + q) * LDX + c] = (f16)v[q];
    }
    __syncthreads();

    const int w = tid >> 5, lane = tid & 31;
    const int m = lane & 15, half = lane >> 4;

    for (int mat = 0; mat < 3; ++mat) {
        const float* Wm = (mat == 0) ? th_w : (mat == 1) ? ph_w : gg_w;
        const float* Bm = (mat == 0) ? th_b : (mat == 1) ? ph_b : gg_b;

        v8f acc[4];
#pragma unroll
        for (int pt = 0; pt < 4; ++pt) acc[pt] = vzero();

#pragma unroll
        for (int cc = 0; cc < 8; ++cc) {
            v16h a = load_a16x32_f32(Wm + (size_t)(w * 16) * CIN + cc * 32, CIN, m, half);
#pragma unroll
            for (int pt = 0; pt < 4; ++pt) {
                v16h b = load_bT(&xT[(pt * 16) * LDX + cc * 32], LDX, m, half);
                acc[pt] = wmma_f16(a, b, acc[pt]);
            }
        }
        v4f bb0 = *(const v4f*)(Bm + w * 16 + 8 * half);
        v4f bb1 = *(const v4f*)(Bm + w * 16 + 8 * half + 4);

        if (mat < 2) {                    // theta/phi: store transposed [pos][chan]
            f16* dst = ((mat == 0) ? thetaT : phiT) + (size_t)batch * HW * CI;
#pragma unroll
            for (int pt = 0; pt < 4; ++pt) {
                v8h pk;
#pragma unroll
                for (int r = 0; r < 8; ++r)
                    pk[r] = (f16)(acc[pt][r] + ((r < 4) ? bb0[r] : bb1[r - 4]));
                *(v8h*)(dst + (size_t)(p0 + pt * 16 + m) * CI + w * 16 + 8 * half) = pk;
            }
        } else {                          // g: store [chan][pos]
            f16* dst = gH + (size_t)batch * CI * HW;
#pragma unroll
            for (int pt = 0; pt < 4; ++pt)
#pragma unroll
                for (int r = 0; r < 8; ++r)
                    dst[(size_t)(w * 16 + 8 * half + r) * HW + p0 + pt * 16 + m] =
                        (f16)(acc[pt][r] + ((r < 4) ? bb0[r] : bb1[r - 4]));
        }
    }
}

// ---------------------------------------------------------------------------
// Kernel 2: per-row softmax stats.  s[i,j] = sum_c phi[c,i]*theta[c,j].
// Wave w owns i-tile (16 rows); A fragments are loop-invariant (hoisted).
// Each lane keeps ONLINE (m,Z) partials for its 8 rows over its j subset;
// cross-lane merge once at the end via LDS.  Writes m_i and 1/Z_i.
// ---------------------------------------------------------------------------
__global__ void __launch_bounds__(256) stats_kernel(
    const f16* __restrict__ thetaT, const f16* __restrict__ phiT,
    float* __restrict__ mrow, float* __restrict__ zinv)
{
    __shared__ float mlds[128 * 16];
    __shared__ float zlds[128 * 16];

    const int bx = blockIdx.x;
    const int batch = bx >> 5, ib = bx & 31;
    const int i0 = ib * 128;
    const f16* phT = phiT   + (size_t)batch * HW * CI;
    const f16* thT = thetaT + (size_t)batch * HW * CI;

    const int tid = threadIdx.x, w = tid >> 5, lane = tid & 31;
    const int m = lane & 15, half = lane >> 4;

    v16h afr[4];
#pragma unroll
    for (int cc = 0; cc < 4; ++cc)
        afr[cc] = load_a16x32(phT + (size_t)(i0 + w * 16) * CI + cc * 32, CI, m, half);

    float m8[8], z8[8];
#pragma unroll
    for (int r = 0; r < 8; ++r) { m8[r] = -3.0e38f; z8[r] = 0.0f; }

    for (int jt = 0; jt < 256; ++jt) {
        v8f s = vzero();
#pragma unroll
        for (int cc = 0; cc < 4; ++cc) {
            v16h b = load_bT(thT + (size_t)(jt * 16) * CI + cc * 32, CI, m, half);
            s = wmma_f16(afr[cc], b, s);
        }
#pragma unroll
        for (int r = 0; r < 8; ++r) {
            float v  = s[r];
            float mn = fmaxf(m8[r], v);
            z8[r] = z8[r] * __expf(m8[r] - mn) + __expf(v - mn);
            m8[r] = mn;
        }
    }
#pragma unroll
    for (int r = 0; r < 8; ++r) {
        int row = w * 16 + 8 * half + r;
        mlds[row * 16 + m] = m8[r];
        zlds[row * 16 + m] = z8[r];
    }
    __syncthreads();
    if (tid < 128) {
        float mm = -3.0e38f;
        for (int l = 0; l < 16; ++l) mm = fmaxf(mm, mlds[tid * 16 + l]);
        float zz = 0.0f;
        for (int l = 0; l < 16; ++l) zz += zlds[tid * 16 + l] * __expf(mlds[tid * 16 + l] - mm);
        mrow[(size_t)batch * HW + i0 + tid] = mm;
        zinv[(size_t)batch * HW + i0 + tid] = 1.0f / zz;
    }
}

// ---------------------------------------------------------------------------
// Kernel 3: fold 1/Z_i into g:  gS[n][c][i] = gH[n][c][i] * zinv[n][i]
// ---------------------------------------------------------------------------
__global__ void __launch_bounds__(256) gscale_kernel(
    const f16* __restrict__ gH, const float* __restrict__ zinv, f16* __restrict__ gS)
{
    const size_t base = ((size_t)blockIdx.x * blockDim.x + threadIdx.x) * 8;
    const int i = (int)(base & (HW - 1));
    const int n = (int)(base >> 19);                // / (128*4096)
    v8h g = *(const v8h*)(gH + base);
    const float* zp = zinv + (size_t)n * HW + i;
    v8h o;
#pragma unroll
    for (int q = 0; q < 8; ++q) o[q] = (f16)((float)g[q] * zp[q]);
    *(v8h*)(gS + base) = o;
}

// ---------------------------------------------------------------------------
// Kernel 4: flash pass 2 + fused output projection + residual.
// Block: one batch x 128-j block. Loop i-chunks of 32:
//   S-phase:  wave w computes S for j-tile w (theta B-frags hoisted), applies
//             exp(s - m_i), stores P^T [j][i] f16 to LDS.
//   att-phase: each wave contracts its 16-channel slab of g' with P (B from LDS).
// Then att^T -> LDS and 256x128 GEMM with w_w, + w_b + x residual -> out.
// ---------------------------------------------------------------------------
__global__ void __launch_bounds__(256) attn_kernel(
    const f16* __restrict__ thetaT, const f16* __restrict__ phiT,
    const f16* __restrict__ gS, const float* __restrict__ mrow,
    const float* __restrict__ x, const float* __restrict__ w_w,
    const float* __restrict__ w_b, float* __restrict__ out)
{
    constexpr int LDP = 40;                 // P^T row stride (halves), 80B (16B-mult)
    constexpr int LDA = 136;                // att^T row stride (halves), 272B
    __shared__ f16 PT[128 * LDP];           // [j_local][i_local 0..31]  (10.0 KB)
    __shared__ f16 attT[128 * LDA];         // [j_local][k 0..127]       (34.0 KB)

    const int bx = blockIdx.x;
    const int batch = bx >> 5, jb = bx & 31;
    const int j0 = jb * 128;
    const f16* thT  = thetaT + (size_t)batch * HW * CI;
    const f16* phT  = phiT   + (size_t)batch * HW * CI;
    const f16* gSb  = gS     + (size_t)batch * CI * HW;
    const float* mr = mrow   + (size_t)batch * HW;

    const int tid = threadIdx.x, w = tid >> 5, lane = tid & 31;
    const int m = lane & 15, half = lane >> 4;

    // theta B fragments for j-tile w: loop-invariant over all i -> hoist.
    v16h bth[4];
#pragma unroll
    for (int cc = 0; cc < 4; ++cc)
        bth[cc] = load_bT(thT + (size_t)(j0 + w * 16) * CI + cc * 32, CI, m, half);

    v8f acc[8];
#pragma unroll
    for (int jt = 0; jt < 8; ++jt) acc[jt] = vzero();

    for (int ic = 0; ic < 128; ++ic) {
        const int i0 = ic * 32;
        // ---- S phase: P^T[j = w*16 + n][i 0..31] ----
#pragma unroll
        for (int isub = 0; isub < 2; ++isub) {
            v8f s = vzero();
#pragma unroll
            for (int cc = 0; cc < 4; ++cc) {
                v16h a = load_a16x32(phT + (size_t)(i0 + isub * 16) * CI + cc * 32,
                                     CI, m, half);
                s = wmma_f16(a, bth[cc], s);
            }
            v4f m0 = *(const v4f*)(mr + i0 + isub * 16 + 8 * half);
            v4f m1 = *(const v4f*)(mr + i0 + isub * 16 + 8 * half + 4);
            v8h pk;
#pragma unroll
            for (int r = 0; r < 8; ++r)
                pk[r] = (f16)__expf(s[r] - ((r < 4) ? m0[r] : m1[r - 4]));
            *(v8h*)(&PT[(w * 16 + m) * LDP + isub * 16 + 8 * half]) = pk;
        }
        __syncthreads();
        // ---- att phase: acc[c-tile w][jt] += g'[c][i-chunk] x P[i][j] ----
        v16h ag = load_a16x32(gSb + (size_t)(w * 16) * HW + i0, HW, m, half);
#pragma unroll
        for (int jt = 0; jt < 8; ++jt) {
            v16h bp = load_bT(&PT[(jt * 16) * LDP], LDP, m, half);
            acc[jt] = wmma_f16(ag, bp, acc[jt]);
        }
        __syncthreads();
    }

    // att^T -> LDS [j][k] f16
#pragma unroll
    for (int jt = 0; jt < 8; ++jt) {
        v8h pk;
#pragma unroll
        for (int r = 0; r < 8; ++r) pk[r] = (f16)acc[jt][r];
        *(v8h*)(&attT[(jt * 16 + m) * LDA + w * 16 + 8 * half]) = pk;
    }
    __syncthreads();

    // Final projection: out[cout][p] = x + w_w[cout][:] @ att[:, p] + w_b[cout]
    const float* xb = x   + (size_t)batch * CIN * HW;
    float*       ob = out + (size_t)batch * CIN * HW;
#pragma unroll
    for (int t2 = 0; t2 < 2; ++t2) {
        const int c0 = (2 * w + t2) * 16;
        v8f a2[8];
#pragma unroll
        for (int pt = 0; pt < 8; ++pt) a2[pt] = vzero();
#pragma unroll
        for (int kc = 0; kc < 4; ++kc) {
            v16h a = load_a16x32_f32(w_w + (size_t)c0 * CI + kc * 32, CI, m, half);
#pragma unroll
            for (int pt = 0; pt < 8; ++pt) {
                v16h b = load_bT(&attT[(pt * 16) * LDA + kc * 32], LDA, m, half);
                a2[pt] = wmma_f16(a, b, a2[pt]);
            }
        }
        v4f b0 = *(const v4f*)(w_b + c0 + 8 * half);
        v4f b1 = *(const v4f*)(w_b + c0 + 8 * half + 4);
#pragma unroll
        for (int pt = 0; pt < 8; ++pt)
#pragma unroll
            for (int r = 0; r < 8; ++r) {
                const int cout = c0 + 8 * half + r;
                const size_t idx = (size_t)cout * HW + j0 + pt * 16 + m;
                ob[idx] = xb[idx] + a2[pt][r] + ((r < 4) ? b0[r] : b1[r - 4]);
            }
    }
}

// ---------------------------------------------------------------------------
// Host launcher.  Workspace layout (bytes):
//   thetaT 0..4Mi | phiT 4..8Mi | gH 8..12Mi | gS 12..16Mi | m 16Mi+ | zinv
// Total ~16.2 MiB (L2-resident on MI455X's 192 MB L2).
// ---------------------------------------------------------------------------
extern "C" void kernel_launch(void* const* d_in, const int* in_sizes, int n_in,
                              void* d_out, int out_size, void* d_ws, size_t ws_size,
                              hipStream_t stream)
{
    (void)in_sizes; (void)n_in; (void)out_size; (void)ws_size;
    const float* x    = (const float*)d_in[0];
    const float* th_w = (const float*)d_in[1];
    const float* th_b = (const float*)d_in[2];
    const float* ph_w = (const float*)d_in[3];
    const float* ph_b = (const float*)d_in[4];
    const float* g_w  = (const float*)d_in[5];
    const float* g_b  = (const float*)d_in[6];
    const float* w_w  = (const float*)d_in[7];
    const float* w_b  = (const float*)d_in[8];
    float* out = (float*)d_out;

    char* ws = (char*)d_ws;
    f16*   thetaT = (f16*)(ws);
    f16*   phiT   = (f16*)(ws + (size_t)4  * 1024 * 1024);
    f16*   gH     = (f16*)(ws + (size_t)8  * 1024 * 1024);
    f16*   gS     = (f16*)(ws + (size_t)12 * 1024 * 1024);
    float* mrow   = (float*)(ws + (size_t)16 * 1024 * 1024);
    float* zinv   = (float*)(ws + (size_t)16 * 1024 * 1024 + 65536);

    proj_kernel<<<N_ * 64, 256, 0, stream>>>(x, th_w, th_b, ph_w, ph_b, g_w, g_b,
                                             thetaT, phiT, gH);
    stats_kernel<<<N_ * 32, 256, 0, stream>>>(thetaT, phiT, mrow, zinv);
    gscale_kernel<<<(N_ * CI * HW) / (256 * 8), 256, 0, stream>>>(gH, zinv, gS);
    attn_kernel<<<N_ * 32, 256, 0, stream>>>(thetaT, phiT, gS, mrow, x, w_w, w_b, out);
}
